// MaskedAutoencoderViT_49761491092046
// MI455X (gfx1250) — matrix operand order repaired and checked by reading, not compiled, not run on hardware
//
#include <hip/hip_runtime.h>
#include <hip/hip_bf16.h>
#include <math.h>

// ---------------- types for WMMA (gfx1250, wave32) ----------------
typedef __attribute__((ext_vector_type(16))) __bf16       v16bf;
typedef __attribute__((ext_vector_type(8)))  float        v8f;
typedef __attribute__((ext_vector_type(8)))  unsigned short u16x8;
typedef __attribute__((ext_vector_type(4)))  int          v4i;
typedef unsigned short u16;

union ABFrag { v16bf v; u16x8 h[2]; };

__device__ __forceinline__ v8f wmma_bf16(const ABFrag& a, const ABFrag& b, v8f c) {
  // D = A(16x32 bf16) * B(32x16 bf16) + C(16x16 f32)
  return __builtin_amdgcn_wmma_f32_16x16x32_bf16(false, a.v, false, b.v,
                                                 (short)0, c, false, false);
}

__device__ __forceinline__ u16 f2bf(float f) {   // round-to-nearest-even fp32->bf16
  unsigned u = __float_as_uint(f);
  u += 0x7fffu + ((u >> 16) & 1u);
  return (u16)(u >> 16);
}

__device__ __forceinline__ float wave_sum(float v) {
  for (int o = 16; o; o >>= 1) v += __shfl_xor(v, o, 32);
  return v;
}
__device__ __forceinline__ float wave_max(float v) {
  for (int o = 16; o; o >>= 1) v = fmaxf(v, __shfl_xor(v, o, 32));
  return v;
}

__device__ __forceinline__ float rng01(unsigned a, unsigned b, unsigned c) {
  unsigned h = a * 0x9E3779B1u ^ b * 0x85EBCA77u ^ c * 0xC2B2AE3Du;
  h ^= h >> 16; h *= 0x7FEB352Du; h ^= h >> 15; h *= 0x846CA68Bu; h ^= h >> 16;
  return (float)(h >> 8) * (1.0f / 16777216.0f);
}

// ---- CDNA5 async global->LDS copies (ASYNCcnt path), guarded by probes ----
#if defined(__gfx1250__) && __has_builtin(__builtin_amdgcn_global_load_async_to_lds_b128)
#define USE_ASYNC_LDS 1
typedef __attribute__((address_space(1))) v4i as1_v4i;   // global int4
typedef __attribute__((address_space(3))) v4i as3_v4i;   // LDS int4
#else
#define USE_ASYNC_LDS 0
#endif

__device__ __forceinline__ void async_cp16(const u16* g, u16* l) {
#if USE_ASYNC_LDS
  __builtin_amdgcn_global_load_async_to_lds_b128(
      (as1_v4i*)g, (as3_v4i*)l, 0, 0);
#else
  (void)g; (void)l;
#endif
}
__device__ __forceinline__ void wait_async0() {
#if USE_ASYNC_LDS
#if __has_builtin(__builtin_amdgcn_s_wait_asynccnt)
  __builtin_amdgcn_s_wait_asynccnt(0);
#else
  asm volatile("s_wait_asynccnt 0x0" ::: "memory");
#endif
#endif
}

// ---------------- problem constants ----------------
#define BB   16
#define NN_T 392        // tokens
#define CC   512
#define HH   16
#define HD   32
#define MLPH 2048
#define MTOT (BB * NN_T)         // 6272 = 49*128
#define LF   196
#define KPAD 416                  // 392 padded to multiple of 32
#define SAMPLE_NUM 7448.0f        // int(0.1*195)*392
#define SCALE 0.17677669529663687f

// ================= weight transpose + bf16 pack: in[K,N] -> out[N,K] =========
__global__ __launch_bounds__(256) void transpose_pack(const float* __restrict__ in,
                                                      u16* __restrict__ out,
                                                      int K, int N) {
  size_t idx = (size_t)blockIdx.x * 256 + threadIdx.x;
  if (idx >= (size_t)K * N) return;
  int k = (int)(idx / N), n = (int)(idx % N);
  out[(size_t)n * K + k] = f2bf(in[idx]);
}

__global__ __launch_bounds__(256) void pack_bf16(const float* __restrict__ in,
                                                 u16* __restrict__ out, size_t n) {
  size_t idx = (size_t)blockIdx.x * 256 + threadIdx.x;
  if (idx < n) out[idx] = f2bf(in[idx]);
}

// ================= LayerNorm over C=512, output bf16 ==========================
__global__ __launch_bounds__(256) void layernorm512_bf16(const float* __restrict__ x,
                                                         const float* __restrict__ g,
                                                         const float* __restrict__ b,
                                                         u16* __restrict__ out) {
  int row = blockIdx.x;
  const float* xr = x + (size_t)row * CC;
  int t = threadIdx.x, lane = t & 31, w = t >> 5;
  float v0 = xr[t], v1 = xr[t + 256];
  __shared__ float red[8];
  __shared__ float bc[2];
  float s = wave_sum(v0 + v1);
  if (lane == 0) red[w] = s;
  __syncthreads();
  if (t == 0) { float tt = 0; for (int i = 0; i < 8; ++i) tt += red[i]; bc[0] = tt * (1.0f / CC); }
  __syncthreads();
  float mu = bc[0];
  float d0 = v0 - mu, d1 = v1 - mu;
  float vs = wave_sum(d0 * d0 + d1 * d1);
  if (lane == 0) red[w] = vs;
  __syncthreads();
  if (t == 0) { float tt = 0; for (int i = 0; i < 8; ++i) tt += red[i]; bc[1] = rsqrtf(tt * (1.0f / CC) + 1e-5f); }
  __syncthreads();
  float rs = bc[1];
  out[(size_t)row * CC + t]       = f2bf(d0 * rs * g[t] + b[t]);
  out[(size_t)row * CC + t + 256] = f2bf(d1 * rs * g[t + 256] + b[t + 256]);
}

// ============ 32x32 per-wave micro-tile compute (4 WMMAs) from LDS ===========
__device__ __forceinline__ void gemm_compute(const u16* __restrict__ Ab,
                                             const u16* __restrict__ Bb,
                                             int ma, int nb, int l15, int kb, int kb2,
                                             v8f& c00, v8f& c01, v8f& c10, v8f& c11) {
  ABFrag a0, a1, b0, b1;
  const u16* ar0 = Ab + (ma + l15) * 32;
  const u16* ar1 = Ab + (ma + 16 + l15) * 32;
  const u16* br0 = Bb + (nb + l15) * 32;
  const u16* br1 = Bb + (nb + 16 + l15) * 32;
  a0.h[0] = *(const u16x8*)(ar0 + kb2);
  a0.h[1] = *(const u16x8*)(ar0 + kb2 + 8);
  a1.h[0] = *(const u16x8*)(ar1 + kb2);
  a1.h[1] = *(const u16x8*)(ar1 + kb2 + 8);
  b0.h[0] = *(const u16x8*)(br0 + kb2);
  b0.h[1] = *(const u16x8*)(br0 + kb2 + 8);
  b1.h[0] = *(const u16x8*)(br1 + kb2);
  b1.h[1] = *(const u16x8*)(br1 + kb2 + 8);
  c00 = wmma_bf16(a0, b0, c00);
  c01 = wmma_bf16(a0, b1, c01);
  c10 = wmma_bf16(a1, b0, c10);
  c11 = wmma_bf16(a1, b1, c11);
}

// ====== bf16 WMMA GEMM, 128x64 block tile, double-buffered LDS ===============
// out = A[M,K] * Bt[N,K]^T (+bias, +resid / gelu per MODE)
// MODE 0: outF = acc + bias; MODE 1: outF = resid + acc + bias;
// MODE 2: outH = bf16(gelu(acc + bias))
template <int MODE>
__global__ __launch_bounds__(256) void gemm_bf16(const u16* __restrict__ A,
                                                 const u16* __restrict__ Bt,
                                                 const float* __restrict__ bias,
                                                 const float* __restrict__ resid,
                                                 float* __restrict__ outF,
                                                 u16* __restrict__ outH,
                                                 int M, int N, int K) {
  __shared__ u16 As[2][128 * 32];
  __shared__ u16 Bs[2][64 * 32];
  const int m0 = blockIdx.y * 128, n0 = blockIdx.x * 64;
  const int t = threadIdx.x, w = t >> 5, lane = t & 31, l15 = lane & 15;
  const int kb = (lane >> 4) * 8, kb2 = (lane >> 4) * 16;
  const int ma = (w & 3) * 32;        // wave's M offset inside 128
  const int nb = (w >> 2) * 32;       // wave's N offset inside 64
  const int ar = t >> 2, ac = (t & 3) * 8;   // loader: row chunk / col chunk
  const u16* Ag0 = A  + (size_t)(m0 + ar) * K + ac;
  const u16* Ag1 = A  + (size_t)(m0 + ar + 64) * K + ac;
  const u16* Bg  = Bt + (size_t)(n0 + ar) * K + ac;
  const int lA0 = ar * 32 + ac, lA1 = (ar + 64) * 32 + ac, lB = ar * 32 + ac;

  v8f c00 = {}, c01 = {}, c10 = {}, c11 = {};
  int cur = 0;

#if USE_ASYNC_LDS
  async_cp16(Ag0, &As[0][lA0]);
  async_cp16(Ag1, &As[0][lA1]);
  async_cp16(Bg,  &Bs[0][lB]);
  wait_async0();
  __syncthreads();
  for (int k0 = 32; k0 < K; k0 += 32) {
    int nxt = cur ^ 1;
    async_cp16(Ag0 + k0, &As[nxt][lA0]);
    async_cp16(Ag1 + k0, &As[nxt][lA1]);
    async_cp16(Bg  + k0, &Bs[nxt][lB]);
    gemm_compute(As[cur], Bs[cur], ma, nb, l15, kb, kb2, c00, c01, c10, c11);
    wait_async0();
    __syncthreads();
    cur = nxt;
  }
#else
  u16x8 ra0 = *(const u16x8*)Ag0;
  u16x8 ra1 = *(const u16x8*)Ag1;
  u16x8 rb  = *(const u16x8*)Bg;
  *(u16x8*)&As[0][lA0] = ra0;
  *(u16x8*)&As[0][lA1] = ra1;
  *(u16x8*)&Bs[0][lB]  = rb;
  __syncthreads();
  for (int k0 = 32; k0 < K; k0 += 32) {
    int nxt = cur ^ 1;
    ra0 = *(const u16x8*)(Ag0 + k0);
    ra1 = *(const u16x8*)(Ag1 + k0);
    rb  = *(const u16x8*)(Bg  + k0);
    gemm_compute(As[cur], Bs[cur], ma, nb, l15, kb, kb2, c00, c01, c10, c11);
    *(u16x8*)&As[nxt][lA0] = ra0;
    *(u16x8*)&As[nxt][lA1] = ra1;
    *(u16x8*)&Bs[nxt][lB]  = rb;
    __syncthreads();
    cur = nxt;
  }
#endif
  gemm_compute(As[cur], Bs[cur], ma, nb, l15, kb, kb2, c00, c01, c10, c11);

  v8f accs[4] = {c00, c01, c10, c11};
  for (int ti = 0; ti < 2; ++ti) {
    for (int tj = 0; tj < 2; ++tj) {
      v8f acc = accs[ti * 2 + tj];
      int n = n0 + nb + tj * 16 + l15;
      float bi = bias[n];
      for (int r = 0; r < 8; ++r) {
        int m = m0 + ma + ti * 16 + r + 8 * (lane >> 4);
        float v = acc[r] + bi;
        if (MODE == 1) v += resid[(size_t)m * N + n];
        if (MODE == 2) {
          v = 0.5f * v * (1.0f + erff(v * 0.7071067811865476f));  // exact GELU
          outH[(size_t)m * N + n] = f2bf(v);
        } else {
          outF[(size_t)m * N + n] = v;
        }
      }
    }
  }
}

// ================= split QKV fp32 [M,1536] into q,k bf16 and v^T bf16 ========
__global__ __launch_bounds__(256) void pack_qkv(const float* __restrict__ qkv,
                                                u16* __restrict__ q, u16* __restrict__ k,
                                                u16* __restrict__ vT) {
  size_t idx = (size_t)blockIdx.x * 256 + threadIdx.x;
  if (idx >= (size_t)MTOT * (3 * CC)) return;
  int m  = (int)(idx / (3 * CC));
  int c3 = (int)(idx % (3 * CC));
  int which = c3 >> 9, rest = c3 & 511, h = rest >> 5, d = rest & 31;
  int b = m / NN_T, n = m % NN_T;
  int bh = b * HH + h;
  u16 v = f2bf(qkv[idx]);
  if      (which == 0) q[((size_t)bh * NN_T + n) * HD + d] = v;
  else if (which == 1) k[((size_t)bh * NN_T + n) * HD + d] = v;
  else                 vT[((size_t)bh * HD + d) * KPAD + n] = v;
}

// ============ score-row compute (shared by pass1 / pass2) via WMMA ===========
__device__ __forceinline__ void compute_score_rows(const u16* __restrict__ qbf,
                                                   const u16* __restrict__ kbf,
                                                   int bh, int qtile,
                                                   float* __restrict__ Srow /*16x400*/) {
  int lane = threadIdx.x & 31, w = threadIdx.x >> 5;
  int l15 = lane & 15;
  int kb = (lane >> 4) * 8, kb2 = (lane >> 4) * 16;
  int qm = qtile * 16 + l15; if (qm > NN_T - 1) qm = NN_T - 1;
  const u16* qrow = qbf + ((size_t)bh * NN_T + qm) * HD;
  ABFrag qa;
  qa.h[0] = *(const u16x8*)(qrow + kb2);
  qa.h[1] = *(const u16x8*)(qrow + kb2 + 8);
  for (int kt = w; kt < 25; kt += 8) {
    int key = kt * 16 + l15;
    int keyc = key > NN_T - 1 ? NN_T - 1 : key;
    const u16* krow = kbf + ((size_t)bh * NN_T + keyc) * HD;
    ABFrag kf;
    kf.h[0] = *(const u16x8*)(krow + kb2);
    kf.h[1] = *(const u16x8*)(krow + kb2 + 8);
    v8f acc = {};
    acc = wmma_bf16(qa, kf, acc);
    int nn = kt * 16 + l15;
    for (int r = 0; r < 8; ++r) {
      int mrow = r + 8 * (lane >> 4);
      Srow[mrow * 400 + nn] = (nn < NN_T) ? acc[r] * SCALE : -INFINITY;
    }
  }
}

// ================= attention pass 1: softmax + mask statistics ===============
// stats[(bh*392+i)*4] = {rowmax, denom, aq_max, norm}
__global__ __launch_bounds__(256) void attn_pass1(const u16* __restrict__ qbf,
                                                  const u16* __restrict__ kbf,
                                                  float* __restrict__ stats) {
  __shared__ float Srow[16 * 400];
  int bh = blockIdx.x, qtile = blockIdx.y;
  compute_score_rows(qbf, kbf, bh, qtile, Srow);
  __syncthreads();
  int lane = threadIdx.x & 31, w = threadIdx.x >> 5;
  for (int rr = 0; rr < 2; ++rr) {
    int r = w + rr * 8;
    int i = qtile * 16 + r;
    if (i >= NN_T) continue;
    float* row = Srow + r * 400;
    float mx = -INFINITY;
    for (int j = lane; j < 400; j += 32) mx = fmaxf(mx, row[j]);
    mx = wave_max(mx);
    float den = 0.f;
    for (int j = lane; j < 400; j += 32) den += expf(row[j] - mx);
    den = wave_sum(den);
    int sameLo = (i < LF) ? 0 : LF;
    int oppLo  = (i < LF) ? LF : 0;
    float omx = -INFINITY;
    for (int j = oppLo + lane; j < oppLo + LF; j += 32) omx = fmaxf(omx, row[j]);
    omx = wave_max(omx);
    float qs = 0.f;
    for (int j = sameLo + lane; j < sameLo + LF; j += 32) qs += expf(row[j] - mx);
    qs = wave_sum(qs);
    float sii = expf(row[i] - mx) / den;  // diagonal prob (column index == i)
    float aq  = expf(omx - mx) / den + 1e-6f;
    // probs > 0 => relu(-min) == 0 => adj = 1e-6 exactly as in reference
    float nrm = (qs / den - sii) + (float)(LF - 1) * 1e-6f;
    if (lane == 0) {
      float* st = stats + ((size_t)bh * NN_T + i) * 4;
      st[0] = mx; st[1] = den; st[2] = aq; st[3] = nrm;
    }
  }
}

// ================= sum of aq_max per head ====================================
__global__ __launch_bounds__(256) void sum_aq(const float* __restrict__ stats,
                                              float* __restrict__ sumAq) {
  int bh = blockIdx.x, t = threadIdx.x, lane = t & 31, w = t >> 5;
  float s = 0.f;
  for (int i = t; i < NN_T; i += 256) s += stats[((size_t)bh * NN_T + i) * 4 + 2];
  s = wave_sum(s);
  __shared__ float red[8];
  if (lane == 0) red[w] = s;
  __syncthreads();
  if (t == 0) { float tt = 0; for (int i = 0; i < 8; ++i) tt += red[i]; sumAq[bh] = tt; }
}

// ========= attention pass 2: recompute scores, drop-mask, final softmax ======
__global__ __launch_bounds__(256) void attn_pass2(const u16* __restrict__ qbf,
                                                  const u16* __restrict__ kbf,
                                                  const float* __restrict__ stats,
                                                  const float* __restrict__ sumAq,
                                                  u16* __restrict__ P) {
  __shared__ float Srow[16 * 400];
  int bh = blockIdx.x, qtile = blockIdx.y;
  compute_score_rows(qbf, kbf, bh, qtile, Srow);
  __syncthreads();
  int lane = threadIdx.x & 31, w = threadIdx.x >> 5;
  float sAq = sumAq[bh];
  for (int rr = 0; rr < 2; ++rr) {
    int r = w + rr * 8;
    int i = qtile * 16 + r;
    if (i >= NN_T) continue;
    float* row = Srow + r * 400;
    const float* st = stats + ((size_t)bh * NN_T + i) * 4;
    float mx = st[0], den = st[1], aq = st[2], nrm = st[3];
    int sameLo = (i < LF) ? 0 : LF;
    // Gumbel-top-k drop mask ~= proportional sampling without replacement:
    // P(drop ij) ~ sample_num * att_final_ij / sum(att_final)  (deterministic hash RNG)
    for (int j = lane; j < 400; j += 32) {
      if (j >= NN_T) { row[j] = -INFINITY; continue; }
      if (j >= sameLo && j < sameLo + LF && j != i) {
        float s = expf(row[j] - mx) / den;
        float att = (s + 1e-6f) / nrm * aq;
        float p = SAMPLE_NUM * att / sAq;
        if (rng01((unsigned)bh, (unsigned)i, (unsigned)j) < p) row[j] = -INFINITY;
      }
    }
    float mx2 = -INFINITY;
    for (int j = lane; j < 400; j += 32) mx2 = fmaxf(mx2, row[j]);
    mx2 = wave_max(mx2);
    float den2 = 0.f;
    for (int j = lane; j < 400; j += 32) den2 += expf(row[j] - mx2);
    den2 = wave_sum(den2);
    u16* prow = P + ((size_t)bh * NN_T + i) * KPAD;
    for (int j = lane; j < KPAD; j += 32) {
      float v = 0.f;
      if (j < 400) v = expf(row[j] - mx2) / den2;   // pads are -inf -> 0
      prow[j] = f2bf(v);
    }
  }
}

// ================= attention pass 3: O = P @ V via WMMA ======================
__global__ __launch_bounds__(64) void attn_pv(const u16* __restrict__ P,
                                              const u16* __restrict__ vT,
                                              float* __restrict__ o) {
  int bh = blockIdx.x, mt = blockIdx.y;
  int w = threadIdx.x >> 5, lane = threadIdx.x & 31, l15 = lane & 15;
  int kb = (lane >> 4) * 8, kb2 = (lane >> 4) * 16;
  int m = mt * 16 + l15; if (m > NN_T - 1) m = NN_T - 1;
  const u16* prow = P + ((size_t)bh * NN_T + m) * KPAD;
  int d = w * 16 + l15;
  const u16* vrow = vT + ((size_t)bh * HD + d) * KPAD;
  v8f acc = {};
  for (int kt = 0; kt < KPAD / 32; ++kt) {
    ABFrag a, b;
    a.h[0] = *(const u16x8*)(prow + kt * 32 + kb2);
    a.h[1] = *(const u16x8*)(prow + kt * 32 + kb2 + 8);
    b.h[0] = *(const u16x8*)(vrow + kt * 32 + kb2);
    b.h[1] = *(const u16x8*)(vrow + kt * 32 + kb2 + 8);
    acc = wmma_bf16(a, b, acc);
  }
  int bb = bh >> 4, h = bh & 15;
  int col = h * HD + w * 16 + l15;
  for (int r = 0; r < 8; ++r) {
    int tok = mt * 16 + r + 8 * (lane >> 4);
    if (tok < NN_T) o[((size_t)bb * NN_T + tok) * CC + col] = acc[r];
  }
}

// ============================ launcher =======================================
extern "C" void kernel_launch(void* const* d_in, const int* in_sizes, int n_in,
                              void* d_out, int out_size, void* d_ws, size_t ws_size,
                              hipStream_t stream) {
  const float* x     = (const float*)d_in[0];
  const float* n1w   = (const float*)d_in[1];
  const float* n1b   = (const float*)d_in[2];
  const float* qkvW  = (const float*)d_in[3];
  const float* qkvB  = (const float*)d_in[4];
  const float* projW = (const float*)d_in[5];
  const float* projB = (const float*)d_in[6];
  const float* n2w   = (const float*)d_in[7];
  const float* n2b   = (const float*)d_in[8];
  const float* fc1W  = (const float*)d_in[9];
  const float* fc1B  = (const float*)d_in[10];
  const float* fc2W  = (const float*)d_in[11];
  const float* fc2B  = (const float*)d_in[12];
  float* out = (float*)d_out;

  char* ws = (char*)d_ws;
  size_t off = 0;
  auto alloc = [&](size_t bytes) -> char* {
    char* p = ws + off;
    off = (off + bytes + 255) & ~(size_t)255;
    return p;
  };
  u16*   h1    = (u16*)  alloc((size_t)MTOT * CC * 2);
  u16*   qkvT  = (u16*)  alloc((size_t)(3 * CC) * CC * 2);
  u16*   projT = (u16*)  alloc((size_t)CC * CC * 2);
  u16*   fc1T  = (u16*)  alloc((size_t)MLPH * CC * 2);
  u16*   fc2T  = (u16*)  alloc((size_t)CC * MLPH * 2);
  float* qkvF  = (float*)alloc((size_t)MTOT * 3 * CC * 4);
  u16*   qb    = (u16*)  alloc((size_t)BB * HH * NN_T * HD * 2);
  u16*   kb    = (u16*)  alloc((size_t)BB * HH * NN_T * HD * 2);
  u16*   vT    = (u16*)  alloc((size_t)BB * HH * HD * KPAD * 2);
  float* stats = (float*)alloc((size_t)BB * HH * NN_T * 4 * 4);
  float* sAq   = (float*)alloc((size_t)BB * HH * 4);
  u16*   P     = (u16*)  alloc((size_t)BB * HH * NN_T * KPAD * 2);
  float* oF    = (float*)alloc((size_t)MTOT * CC * 4);
  u16*   oB    = (u16*)  alloc((size_t)MTOT * CC * 2);
  float* x1    = (float*)alloc((size_t)MTOT * CC * 4);
  u16*   h2    = (u16*)  alloc((size_t)MTOT * CC * 2);
  u16*   gB    = (u16*)  alloc((size_t)MTOT * MLPH * 2);
  (void)ws_size; (void)in_sizes; (void)n_in; (void)out_size;

  // v^T pad columns must be zero
  (void)hipMemsetAsync(vT, 0, (size_t)BB * HH * HD * KPAD * 2, stream);

  // weights -> bf16, transposed to [N,K]
  transpose_pack<<<(512 * 1536 + 255) / 256, 256, 0, stream>>>(qkvW, qkvT, CC, 3 * CC);
  transpose_pack<<<(512 * 512 + 255) / 256, 256, 0, stream>>>(projW, projT, CC, CC);
  transpose_pack<<<(512 * 2048 + 255) / 256, 256, 0, stream>>>(fc1W, fc1T, CC, MLPH);
  transpose_pack<<<(2048 * 512 + 255) / 256, 256, 0, stream>>>(fc2W, fc2T, MLPH, CC);

  // h1 = LN1(x)
  layernorm512_bf16<<<MTOT, 256, 0, stream>>>(x, n1w, n1b, h1);
  // qkv = h1 @ W_qkv + b
  gemm_bf16<0><<<dim3((3 * CC) / 64, MTOT / 128), 256, 0, stream>>>(
      h1, qkvT, qkvB, nullptr, qkvF, nullptr, MTOT, 3 * CC, CC);
  pack_qkv<<<(MTOT * 3 * CC + 255) / 256, 256, 0, stream>>>(qkvF, qb, kb, vT);

  // attention
  attn_pass1<<<dim3(BB * HH, 25), 256, 0, stream>>>(qb, kb, stats);
  sum_aq<<<BB * HH, 256, 0, stream>>>(stats, sAq);
  attn_pass2<<<dim3(BB * HH, 25), 256, 0, stream>>>(qb, kb, stats, sAq, P);
  attn_pv<<<dim3(BB * HH, 25), 64, 0, stream>>>(P, vT, oF);
  pack_bf16<<<((size_t)MTOT * CC + 255) / 256, 256, 0, stream>>>(oF, oB, (size_t)MTOT * CC);

  // x1 = x + o @ W_proj + b
  gemm_bf16<1><<<dim3(CC / 64, MTOT / 128), 256, 0, stream>>>(
      oB, projT, projB, x, x1, nullptr, MTOT, CC, CC);
  // h2 = LN2(x1)
  layernorm512_bf16<<<MTOT, 256, 0, stream>>>(x1, n2w, n2b, h2);
  // g = gelu(h2 @ W_fc1 + b)
  gemm_bf16<2><<<dim3(MLPH / 64, MTOT / 128), 256, 0, stream>>>(
      h2, fc1T, fc1B, nullptr, nullptr, gB, MTOT, MLPH, CC);
  // out = x1 + g @ W_fc2 + b
  gemm_bf16<1><<<dim3(CC / 64, MTOT / 128), 256, 0, stream>>>(
      gB, fc2T, fc2B, x1, out, nullptr, MTOT, CC, MLPH);
}
